// GraphAttention_wjq_73839077752943
// MI455X (gfx1250) — matrix operand order repaired and checked by reading, not compiled, not run on hardware
//
#include <hip/hip_runtime.h>
#include <hip/hip_bf16.h>
#include <math.h>

#define N_NODES 25000
#define N_EDGES 400000
#define E_HID   300
#define R_HID   100
#define OUT_W   700   // 2*E_HID + R_HID
#define N_CHUNK 10    // ceil(300/32)

typedef __attribute__((ext_vector_type(16))) _Float16 v16h;
typedef __attribute__((ext_vector_type(8)))  float    v8f;

__device__ __forceinline__ float lrelu(float f) { return f > 0.0f ? f : 0.01f * f; }

// float -> monotonically ordered uint key (for atomicMax-based segment max)
__device__ __forceinline__ unsigned int f2key(float f) {
    unsigned int u = __float_as_uint(f);
    return (u & 0x80000000u) ? ~u : (u | 0x80000000u);
}
__device__ __forceinline__ float key2f(unsigned int k) {
    unsigned int u = (k & 0x80000000u) ? (k & 0x7FFFFFFFu) : ~k;
    return __uint_as_float(u);
}

// ---------------------------------------------------------------- K0: init
__global__ void k_init(float* out, int out_n, unsigned int* smax, float* ssum) {
    long long i = (long long)blockIdx.x * blockDim.x + threadIdx.x;
    long long stride = (long long)gridDim.x * blockDim.x;
    for (long long t = i; t < out_n; t += stride) out[t] = 0.0f;
    for (long long t = i; t < N_NODES; t += stride) { smax[t] = 0u; ssum[t] = 0.0f; }
}

// ---------------- K0b: bake WMMA B-matrix tiles (per-lane layout) from W
// Bbuf[chunk][lane][16 f16]; lane col = lane&15, khalf = lane>>4,
// element j holds K = chunk*32 + (j<8 ? khalf*8 + j : 16 + khalf*8 + (j-8)) of column col.
// col 0 = W[0:300], col 1 = W[400:700], others = 0; k>=300 -> 0.
__global__ void k_build_b(const float* __restrict__ W, _Float16* __restrict__ Bbuf) {
    int tid = blockIdx.x * blockDim.x + threadIdx.x;   // 0 .. 320
    if (tid >= N_CHUNK * 32) return;
    int c     = tid >> 5;
    int lane  = tid & 31;
    int col   = lane & 15;
    int khalf = lane >> 4;
    _Float16* dst = Bbuf + (size_t)tid * 16;
    for (int j = 0; j < 16; ++j) {
        int k = c * 32 + (j < 8 ? khalf * 8 + j : 16 + khalf * 8 + (j - 8));
        float w = 0.0f;
        if (k < E_HID) {
            if (col == 0)      w = W[k];
            else if (col == 1) w = W[400 + k];
        }
        dst[j] = (_Float16)w;
    }
}

// ------------------------------------------ K1: node scores s1/s3 via WMMA
// One wave per 16-node tile.  D = A(lrelu(x) 16x32 f16) x B(32x16 f16) acc f32.
__global__ __launch_bounds__(32) void k_node_scores(const float* __restrict__ x,
                                                    const _Float16* __restrict__ Bbuf,
                                                    float* __restrict__ s1,
                                                    float* __restrict__ s3) {
    const int tile  = blockIdx.x;
    const int lane  = threadIdx.x;
    const int m     = lane & 15;
    const int khalf = lane >> 4;
    const int row   = tile * 16 + m;
    const bool rowok = row < N_NODES;
    const float* __restrict__ xrow = x + (size_t)row * E_HID;

    v8f acc = {};

    // Chunks 0..8: k in [0,288) — always fully in-bounds; vector loads, no guards.
    for (int c = 0; c < 9; ++c) {
        const int base = c * 32;
        v16h a;
        if (rowok) {
            // group 0: K = base + khalf*8 + 0..7 ; group 1: K = base + 16 + khalf*8 + 0..7
            const float4 p0 = *(const float4*)(xrow + base + khalf * 8);
            const float4 p1 = *(const float4*)(xrow + base + khalf * 8 + 4);
            const float4 p2 = *(const float4*)(xrow + base + 16 + khalf * 8);
            const float4 p3 = *(const float4*)(xrow + base + 16 + khalf * 8 + 4);
            a[0]  = (_Float16)lrelu(p0.x); a[1]  = (_Float16)lrelu(p0.y);
            a[2]  = (_Float16)lrelu(p0.z); a[3]  = (_Float16)lrelu(p0.w);
            a[4]  = (_Float16)lrelu(p1.x); a[5]  = (_Float16)lrelu(p1.y);
            a[6]  = (_Float16)lrelu(p1.z); a[7]  = (_Float16)lrelu(p1.w);
            a[8]  = (_Float16)lrelu(p2.x); a[9]  = (_Float16)lrelu(p2.y);
            a[10] = (_Float16)lrelu(p2.z); a[11] = (_Float16)lrelu(p2.w);
            a[12] = (_Float16)lrelu(p3.x); a[13] = (_Float16)lrelu(p3.y);
            a[14] = (_Float16)lrelu(p3.z); a[15] = (_Float16)lrelu(p3.w);
        } else {
            a = (v16h)(_Float16)0.0f;
        }
        const v16h b = *(const v16h*)(Bbuf + ((size_t)c * 32 + lane) * 16);
        acc = __builtin_amdgcn_wmma_f32_16x16x32_f16(false, a, false, b,
                                                     (short)0, acc, false, false);
    }

    // Chunk 9: k = 288..319; only 288..299 valid. Group 1 (k>=304) is all zero.
    {
        v16h a = (v16h)(_Float16)0.0f;
        if (rowok) {
            if (khalf == 0) {                       // K = 288..295, fully valid
                const float4 p0 = *(const float4*)(xrow + 288);
                const float4 p1 = *(const float4*)(xrow + 292);
                a[0] = (_Float16)lrelu(p0.x); a[1] = (_Float16)lrelu(p0.y);
                a[2] = (_Float16)lrelu(p0.z); a[3] = (_Float16)lrelu(p0.w);
                a[4] = (_Float16)lrelu(p1.x); a[5] = (_Float16)lrelu(p1.y);
                a[6] = (_Float16)lrelu(p1.z); a[7] = (_Float16)lrelu(p1.w);
            } else {                                // K = 296..303 -> only 296..299 valid
                const float4 p0 = *(const float4*)(xrow + 296);
                a[0] = (_Float16)lrelu(p0.x); a[1] = (_Float16)lrelu(p0.y);
                a[2] = (_Float16)lrelu(p0.z); a[3] = (_Float16)lrelu(p0.w);
            }
        }
        const v16h b = *(const v16h*)(Bbuf + ((size_t)9 * 32 + lane) * 16);
        acc = __builtin_amdgcn_wmma_f32_16x16x32_f16(false, a, false, b,
                                                     (short)0, acc, false, false);
    }

    // D layout (f32 16x16): lanes 0-15: N=lane, VGPR j -> M=j ; lanes 16-31: N=lane-16, M=8+j
    const int mbase = tile * 16 + khalf * 8;
    if (m == 0) {                                   // column 0 -> s1
        #pragma unroll
        for (int j = 0; j < 8; ++j)
            if (mbase + j < N_NODES) s1[mbase + j] = acc[j];
    } else if (m == 1) {                            // column 1 -> s3
        #pragma unroll
        for (int j = 0; j < 8; ++j)
            if (mbase + j < N_NODES) s3[mbase + j] = acc[j];
    }
}

// ------------------------------------------------- K2: relation scores sr[r]
__global__ void k_rel_scores(const float* __restrict__ rel_emb,
                             const float* __restrict__ W,
                             float* __restrict__ sr, int n_rels) {
    int wave = (blockIdx.x * blockDim.x + threadIdx.x) >> 5;
    int lane = threadIdx.x & 31;
    if (wave >= n_rels) return;
    float v = 0.0f;
    for (int k = lane; k < R_HID; k += 32)
        v += lrelu(rel_emb[(size_t)wave * R_HID + k]) * W[E_HID + k];
    #pragma unroll
    for (int off = 16; off > 0; off >>= 1) v += __shfl_down(v, off, 32);
    if (lane == 0) sr[wave] = v;
}

// ------------------------------- K3: edge score + segment max (ordered uint)
__global__ void k_edge_score(const int* __restrict__ ei, const int* __restrict__ ej,
                             const int* __restrict__ rel,
                             const float* __restrict__ s1, const float* __restrict__ s3,
                             const float* __restrict__ sr,
                             float* __restrict__ att, unsigned int* __restrict__ smax) {
    int e = blockIdx.x * blockDim.x + threadIdx.x;
    if (e >= N_EDGES) return;
    int dst = ei[e];
    float a = s1[dst] + sr[rel[e]] + s3[ej[e]];
    att[e] = a;
    atomicMax(&smax[dst], f2key(a));
}

// ----------------------------------------- K4: exp(att - max) + segment sum
__global__ void k_edge_exp(const int* __restrict__ ei,
                           float* __restrict__ att,
                           const unsigned int* __restrict__ smax,
                           float* __restrict__ ssum) {
    int e = blockIdx.x * blockDim.x + threadIdx.x;
    if (e >= N_EDGES) return;
    int dst = ei[e];
    float ex = __expf(att[e] - key2f(smax[dst]));
    att[e] = ex;                       // now holds exp value
    atomicAdd(&ssum[dst], ex);
}

// ----------------- K5: weighted scatter-aggregate of cols 300:700, per wave
__global__ void k_aggregate(const int* __restrict__ ei, const int* __restrict__ ej,
                            const int* __restrict__ rel,
                            const float* __restrict__ x,
                            const float* __restrict__ rel_emb,
                            const float* __restrict__ att_exp,
                            const float* __restrict__ ssum,
                            float* __restrict__ out) {
    int e    = (blockIdx.x * blockDim.x + threadIdx.x) >> 5;
    int lane = threadIdx.x & 31;
    if (e >= N_EDGES) return;
    int dst = ei[e];
    int src = ej[e];
    int r   = rel[e];
    float w = att_exp[e] / (ssum[dst] + 1e-16f);

    const float* xr = x + (size_t)src * E_HID;
    __builtin_prefetch(xr + lane, 0, 0);                 // global_prefetch_b8 on gathered row

    float* od = out + (size_t)dst * OUT_W;
    const float* rr = rel_emb + (size_t)r * R_HID;
    for (int k = lane; k < R_HID; k += 32)
        atomicAdd(&od[E_HID + k], w * lrelu(rr[k]));
    for (int k = lane; k < E_HID; k += 32)
        atomicAdd(&od[E_HID + R_HID + k], w * lrelu(xr[k]));
}

// --------------------------- K6: cols 0:300 = S_n * lrelu(x[n]) (no scatter)
__global__ void k_self_cols(const float* __restrict__ x,
                            const float* __restrict__ ssum,
                            float* __restrict__ out) {
    long long i = (long long)blockIdx.x * blockDim.x + threadIdx.x;
    if (i >= (long long)N_NODES * E_HID) return;
    int n = (int)(i / E_HID);
    int k = (int)(i % E_HID);
    float s = ssum[n];
    float scale = s / (s + 1e-16f);    // == sum of normalized softmax weights
    out[(size_t)n * OUT_W + k] = scale * lrelu(x[(size_t)n * E_HID + k]);
}

extern "C" void kernel_launch(void* const* d_in, const int* in_sizes, int n_in,
                              void* d_out, int out_size, void* d_ws, size_t ws_size,
                              hipStream_t stream) {
    const float* x        = (const float*)d_in[0];
    const int*   eidx     = (const int*)d_in[1];     // [2, E] flat
    const int*   rel      = (const int*)d_in[2];
    const float* rel_emb  = (const float*)d_in[3];
    const float* W        = (const float*)d_in[4];
    float* out = (float*)d_out;

    const int* ei = eidx;            // row 0: destination / segment key
    const int* ej = eidx + N_EDGES;  // row 1

    // workspace layout (float units):
    // s1[25000] s3[25000] sr[1000] smax[25000] ssum[25000] att[400000] Bbuf[2560]
    float*        wsf  = (float*)d_ws;
    float*        s1   = wsf;
    float*        s3   = wsf + 25000;
    float*        sr   = wsf + 50000;
    unsigned int* smax = (unsigned int*)(wsf + 51000);
    float*        ssum = wsf + 76000;
    float*        att  = wsf + 101000;
    _Float16*     Bbuf = (_Float16*)(wsf + 501000);  // 10*32*16 f16 = 10 KB, 32B aligned

    // K0: zero output + segment buffers; bake B tiles
    k_init<<<4096, 256, 0, stream>>>(out, out_size, smax, ssum);
    k_build_b<<<2, 256, 0, stream>>>(W, Bbuf);

    // K1: WMMA node scores (one wave per 16-node tile)
    k_node_scores<<<(N_NODES + 15) / 16, 32, 0, stream>>>(x, Bbuf, s1, s3);

    // K2: relation scores (one wave per relation)
    k_rel_scores<<<(1000 * 32 + 255) / 256, 256, 0, stream>>>(rel_emb, W, sr, 1000);

    // K3: edge scores + segment max
    k_edge_score<<<(N_EDGES + 255) / 256, 256, 0, stream>>>(ei, ej, rel, s1, s3, sr, att, smax);

    // K4: exp + segment sum
    k_edge_exp<<<(N_EDGES + 255) / 256, 256, 0, stream>>>(ei, att, smax, ssum);

    // K5: scatter-aggregate cols 300:700 (one wave per edge)
    k_aggregate<<<(N_EDGES * 32 + 255) / 256, 256, 0, stream>>>(ei, ej, rel, x, rel_emb, att, ssum, out);

    // K6: direct write of cols 0:300
    long long selfN = (long long)N_NODES * E_HID;
    k_self_cols<<<(unsigned)((selfN + 255) / 256), 256, 0, stream>>>(x, ssum, out);
}
